// GPTBlock_43894565765722
// MI455X (gfx1250) — matrix operand order repaired and checked by reading, compiled, not run on hardware
//
#include <hip/hip_runtime.h>
#include <hip/hip_bf16.h>

// ---------------------------------------------------------------------------
// GPT block for gfx1250 (MI455X): all GEMMs via v_wmma_f32_16x16x32_f16,
// fp32 accumulate, f16 operands staged through LDS with layouts chosen so
// every WMMA fragment is loaded with two aligned ds_load_b128.
// ---------------------------------------------------------------------------

typedef __attribute__((ext_vector_type(16))) _Float16 v16h;
typedef __attribute__((ext_vector_type(8)))  _Float16 v8h;
typedef __attribute__((ext_vector_type(2)))  _Float16 v2h;
typedef __attribute__((ext_vector_type(8)))  float    v8f;

#define D_MODEL 1024
#define N_HEADS 16
#define HEAD_DIM 64
#define HIDDEN 2880
#define SEQ 2048
#define BATCH 4
#define TOK (BATCH * SEQ)      // 8192
#define H3 (3 * D_MODEL)       // 3072

__device__ __forceinline__ v16h cat8(v8h lo, v8h hi) {
  return __builtin_shufflevector(lo, hi, 0, 1, 2, 3, 4, 5, 6, 7,
                                 8, 9, 10, 11, 12, 13, 14, 15);
}

// ---------------- WMMA fragment loaders (ISA 7.12.2 layouts) ----------------
// A 16x32 f16, row-major LDS tile: lane half selects K-base {0,8}; the lane's
// 16 values are two contiguous 8-half runs -> 2x ds_load_b128.
__device__ __forceinline__ v16h frag_a(const _Float16* base, int stride, int lane) {
  int m  = lane & 15;
  int kb = (lane >> 4) * 8;
  const _Float16* p = base + m * stride + kb;
  return cat8(*(const v8h*)p, *(const v8h*)(p + 16));
}

// B 32x16 f16 from a COLUMN-MAJOR LDS tile (element (k,n) at base[n*stride+k]):
// lane half selects K-base {0,8}; lane's 16 values are two 8-half runs
// K=kb..kb+7 and kb+16..kb+23 (same order as frag_a) -> 2x ds_load_b128.
__device__ __forceinline__ v16h frag_bt(const _Float16* base, int stride, int lane) {
  int n  = lane & 15;
  int kb = (lane >> 4) * 8;
  const _Float16* p = base + n * stride + kb;
  return cat8(*(const v8h*)p, *(const v8h*)(p + 16));
}

__device__ __forceinline__ v8f wmma_f16(v16h a, v16h b, v8f c) {
  return __builtin_amdgcn_wmma_f32_16x16x32_f16(false, a, false, b,
                                                (short)0, c, false, false);
}

// ---------------------------- utility kernels -------------------------------
__global__ __launch_bounds__(256) void cvt_f32_f16(const float* __restrict__ src,
                                                   _Float16* __restrict__ dst,
                                                   int rows, int cols,
                                                   int dstride, int doff) {
  int idx = blockIdx.x * 256 + threadIdx.x;
  if (idx >= rows * cols) return;
  int r = idx / cols, c = idx % cols;
  dst[(size_t)r * dstride + doff + c] = (_Float16)src[idx];
}

__global__ __launch_bounds__(256) void ln_f16(const float* __restrict__ x,
                                              const float* __restrict__ g,
                                              const float* __restrict__ b,
                                              _Float16* __restrict__ out) {
  int row = blockIdx.x;
  int tid = threadIdx.x;
  const float* xr = x + (size_t)row * D_MODEL;
  __shared__ float red[256];
  float s = 0.f;
  for (int c = tid; c < D_MODEL; c += 256) s += xr[c];
  red[tid] = s;
  __syncthreads();
  for (int o = 128; o > 0; o >>= 1) {
    if (tid < o) red[tid] += red[tid + o];
    __syncthreads();
  }
  float mu = red[0] * (1.0f / D_MODEL);
  __syncthreads();
  float v = 0.f;
  for (int c = tid; c < D_MODEL; c += 256) {
    float d = xr[c] - mu;
    v += d * d;
  }
  red[tid] = v;
  __syncthreads();
  for (int o = 128; o > 0; o >>= 1) {
    if (tid < o) red[tid] += red[tid + o];
    __syncthreads();
  }
  float rstd = rsqrtf(red[0] * (1.0f / D_MODEL) + 1e-5f);
  for (int c = tid; c < D_MODEL; c += 256)
    out[(size_t)row * D_MODEL + c] = (_Float16)((xr[c] - mu) * rstd * g[c] + b[c]);
}

__global__ __launch_bounds__(256) void swiglu_f16(_Float16* __restrict__ g1,
                                                  const _Float16* __restrict__ g3,
                                                  int n) {
  int i = blockIdx.x * 256 + threadIdx.x;
  if (i >= n) return;
  float a = (float)g1[i];
  float s = a / (1.0f + __expf(-a));
  g1[i] = (_Float16)(s * (float)g3[i]);
}

// ------------------------------- GEMM ---------------------------------------
// Block tile 128(M) x 64(N); 4 waves, each 64x32 (4x2 tiles of 16x16).
// A staged row-major; B staged TRANSPOSED (column-major) via packed half2.
// EPI 0: f16 output (+optional bias). EPI 1: f32 output (+bias)(+residual).
#define LDA_S 40   // row stride of 128x32 A tile (80B = 5*16)
#define LDT_S 40   // row stride of 64(n)x32(k) transposed B tile

template <int EPI>
__global__ __launch_bounds__(128)
void gemm_f16(const _Float16* __restrict__ A, int lda,
              const _Float16* __restrict__ Bm, int ldb,
              int K,
              const float* __restrict__ bias,
              const float* __restrict__ resid,
              _Float16* __restrict__ Ch, float* __restrict__ Cf, int ldc) {
  __shared__ _Float16 As[128][LDA_S];
  __shared__ _Float16 Bt[64][LDT_S];   // Bt[n][k]

  const int tid  = threadIdx.x;
  const int lane = tid & 31;
  const int wave = tid >> 5;
  const int wr   = wave >> 1;          // 0..1 -> 64-row strip
  const int wc   = wave & 1;           // 0..1 -> 32-col strip
  const int hf   = lane >> 4;
  const int cn   = lane & 15;
  const int m0   = blockIdx.y * 128;
  const int n0   = blockIdx.x * 64;

  v8f acc[4][2] = {};

  for (int k0 = 0; k0 < K; k0 += 32) {
    // stage A: 128 rows x 32 halfs = 512 16B chunks, 4 per thread
#pragma unroll
    for (int i = 0; i < 4; ++i) {
      int idx = tid + 128 * i;
      int r = idx >> 2, c = (idx & 3) * 8;
      *(v8h*)&As[r][c] = *(const v8h*)(A + (size_t)(m0 + r) * lda + k0 + c);
    }
    // stage B transposed: 16 k-pairs x 8 n-groups = 128 chunks, 1 per thread.
    // Each thread reads two global rows (k, k+1) x 8 cols and writes 8 packed
    // half2 (ds_store_b32) into column-major Bt.
    {
      int kp = tid >> 3;            // 0..15
      int ng = tid & 7;             // 0..7
      int k  = 2 * kp;
      const _Float16* gb = Bm + (size_t)(k0 + k) * ldb + n0 + ng * 8;
      v8h r0 = *(const v8h*)gb;
      v8h r1 = *(const v8h*)(gb + ldb);
#pragma unroll
      for (int e = 0; e < 8; ++e) {
        v2h pk;
        pk[0] = r0[e];
        pk[1] = r1[e];
        *(v2h*)&Bt[ng * 8 + e][k] = pk;
      }
    }
    if (k0 + 32 < K) {  // global_prefetch_b8 of next K slab
      __builtin_prefetch(A + (size_t)(m0 + tid) * lda + k0 + 32, 0, 1);
      __builtin_prefetch(Bm + (size_t)(k0 + 32 + (tid & 31)) * ldb + n0, 0, 1);
    }
    __syncthreads();

    v16h af[4], bf[2];
#pragma unroll
    for (int i = 0; i < 4; ++i)
      af[i] = frag_a(&As[wr * 64 + 16 * i][0], LDA_S, lane);
#pragma unroll
    for (int j = 0; j < 2; ++j)
      bf[j] = frag_bt(&Bt[wc * 32 + 16 * j][0], LDT_S, lane);
#pragma unroll
    for (int i = 0; i < 4; ++i)
#pragma unroll
      for (int j = 0; j < 2; ++j)
        acc[i][j] = wmma_f16(af[i], bf[j], acc[i][j]);
    __syncthreads();
  }

  // epilogue: C layout row = 8*half + v, col = lane&15
#pragma unroll
  for (int i = 0; i < 4; ++i) {
#pragma unroll
    for (int j = 0; j < 2; ++j) {
      int rbase = m0 + wr * 64 + 16 * i + 8 * hf;
      int cbase = n0 + wc * 32 + 16 * j + cn;
      float bv = bias ? bias[cbase] : 0.f;
#pragma unroll
      for (int v = 0; v < 8; ++v) {
        float val = acc[i][j][v] + bv;
        size_t off = (size_t)(rbase + v) * ldc + cbase;
        if (EPI == 0) {
          Ch[off] = (_Float16)val;
        } else {
          if (resid) val += resid[off];
          Cf[off] = val;
        }
      }
    }
  }
}

// ---------------------------- attention -------------------------------------
// grid = (SEQ/64, BATCH*N_HEADS); 128 threads (4 waves). Flash-style online
// softmax over 64-key tiles; QK^T and PV through WMMA.
//   Qs, Ps: row-major (A-fragments, contiguous)
//   Ks: row-major [key][d] -> K^T fragments contiguous along d (frag_bt)
//   Vt: transposed [d][key] -> V fragments contiguous along key (frag_bt)
#define ATT_S 72

__global__ __launch_bounds__(128)
void attn_kernel(const _Float16* __restrict__ qkv, _Float16* __restrict__ out) {
  __shared__ _Float16 Qs[64][ATT_S];
  __shared__ _Float16 Ks[64][ATT_S];
  __shared__ _Float16 Vt[64][ATT_S];
  __shared__ _Float16 Ps[64][ATT_S];
  __shared__ float    Ss[64][65];
  __shared__ float    mrow[64], lrow[64], crow[64];

  const int tid  = threadIdx.x;
  const int lane = tid & 31;
  const int w    = tid >> 5;
  const int hf   = lane >> 4;
  const int cn   = lane & 15;
  const int qt   = blockIdx.x;                 // q-tile
  const int bh   = blockIdx.y;
  const int bidx = bh / N_HEADS;
  const int head = bh % N_HEADS;
  const int tb   = bidx * SEQ;                 // token base for this batch
  const int hq   = head * HEAD_DIM;
  const int q0   = qt * 64;

  // load Q tile (64 x 64, row-major)
#pragma unroll
  for (int i = 0; i < 4; ++i) {
    int idx = tid + 128 * i;
    int r = idx >> 3, c = (idx & 7) * 8;
    *(v8h*)&Qs[r][c] = *(const v8h*)(qkv + (size_t)(tb + q0 + r) * H3 + hq + c);
  }
  if (tid < 64) { mrow[tid] = -1e30f; lrow[tid] = 0.f; }
  __syncthreads();

  v8f o[4] = {};  // wave's 16-row strip x 64 cols

  for (int j = 0; j <= qt; ++j) {
    const int kb0 = j * 64;
    // K tile row-major: 64 rows x 8 chunks = 512, 4 per thread
#pragma unroll
    for (int i = 0; i < 4; ++i) {
      int idx = tid + 128 * i;
      int r = idx >> 3, c = (idx & 7) * 8;
      *(v8h*)&Ks[r][c] =
          *(const v8h*)(qkv + (size_t)(tb + kb0 + r) * H3 + hq + D_MODEL + c);
    }
    // V tile transposed: 32 key-pairs x 8 d-groups = 256 chunks, 2 per thread
#pragma unroll
    for (int i = 0; i < 2; ++i) {
      int idx = tid + 128 * i;
      int kp = idx >> 3, ng = idx & 7;
      int key = 2 * kp;
      const _Float16* gv =
          qkv + (size_t)(tb + kb0 + key) * H3 + hq + 2 * D_MODEL + ng * 8;
      v8h r0 = *(const v8h*)gv;
      v8h r1 = *(const v8h*)(gv + H3);
#pragma unroll
      for (int e = 0; e < 8; ++e) {
        v2h pk;
        pk[0] = r0[e];
        pk[1] = r1[e];
        *(v2h*)&Vt[ng * 8 + e][key] = pk;
      }
    }
    __syncthreads();

    // S = Q @ K^T  (wave handles rows 16w..16w+15, all 64 key columns)
    {
      v16h aq[2];
#pragma unroll
      for (int kk = 0; kk < 2; ++kk)
        aq[kk] = frag_a(&Qs[16 * w][kk * 32], ATT_S, lane);
      v8f sacc[4] = {};
#pragma unroll
      for (int jn = 0; jn < 4; ++jn)
#pragma unroll
        for (int kk = 0; kk < 2; ++kk)
          sacc[jn] = wmma_f16(aq[kk],
                              frag_bt(&Ks[16 * jn][kk * 32], ATT_S, lane),
                              sacc[jn]);
#pragma unroll
      for (int jn = 0; jn < 4; ++jn)
#pragma unroll
        for (int v = 0; v < 8; ++v)
          Ss[16 * w + 8 * hf + v][16 * jn + cn] = sacc[jn][v];
    }
    __syncthreads();

    // online softmax per row (threads 0..63 each own one row)
    if (tid < 64) {
      int r = tid;
      int qpos = q0 + r;
      int lim = (j == qt) ? (qpos - kb0 + 1) : 64;  // # of unmasked cols
      float mx = mrow[r];
      for (int c = 0; c < lim; ++c) mx = fmaxf(mx, Ss[r][c] * 0.125f);
      float corr = __expf(mrow[r] - mx);
      float psum = 0.f;
      for (int c = 0; c < 64; c += 2) {
        float p0 = (c < lim) ? __expf(Ss[r][c] * 0.125f - mx) : 0.f;
        float p1 = (c + 1 < lim) ? __expf(Ss[r][c + 1] * 0.125f - mx) : 0.f;
        v2h pk;
        pk[0] = (_Float16)p0;
        pk[1] = (_Float16)p1;
        *(v2h*)&Ps[r][c] = pk;   // packed ds_store_b32
        psum += p0 + p1;
      }
      mrow[r] = mx;
      lrow[r] = lrow[r] * corr + psum;
      crow[r] = corr;
    }
    __syncthreads();

    // rescale O, then O += P @ V
    {
      float cr[8];
#pragma unroll
      for (int v = 0; v < 8; ++v) cr[v] = crow[16 * w + 8 * hf + v];
#pragma unroll
      for (int jn = 0; jn < 4; ++jn)
#pragma unroll
        for (int v = 0; v < 8; ++v) o[jn][v] *= cr[v];

      v16h ap[2];
#pragma unroll
      for (int kk = 0; kk < 2; ++kk)
        ap[kk] = frag_a(&Ps[16 * w][kk * 32], ATT_S, lane);
#pragma unroll
      for (int jn = 0; jn < 4; ++jn)
#pragma unroll
        for (int kk = 0; kk < 2; ++kk)
          o[jn] = wmma_f16(ap[kk],
                           frag_bt(&Vt[16 * jn][kk * 32], ATT_S, lane),
                           o[jn]);
    }
    __syncthreads();
  }

  // normalize by l and store f16 output at [token][head*64 + col]
  float li[8];
#pragma unroll
  for (int v = 0; v < 8; ++v) li[v] = 1.0f / lrow[16 * w + 8 * hf + v];
#pragma unroll
  for (int jn = 0; jn < 4; ++jn)
#pragma unroll
    for (int v = 0; v < 8; ++v) {
      int r = 16 * w + 8 * hf + v;
      out[(size_t)(tb + q0 + r) * D_MODEL + hq + 16 * jn + cn] =
          (_Float16)(o[jn][v] * li[v]);
    }
}

// ------------------------------ launcher ------------------------------------
extern "C" void kernel_launch(void* const* d_in, const int* in_sizes, int n_in,
                              void* d_out, int out_size, void* d_ws, size_t ws_size,
                              hipStream_t stream) {
  (void)in_sizes; (void)n_in; (void)out_size; (void)ws_size;
  const float* x    = (const float*)d_in[0];
  const float* ln1g = (const float*)d_in[1];
  const float* ln1b = (const float*)d_in[2];
  const float* wq   = (const float*)d_in[3];
  const float* bq   = (const float*)d_in[4];
  const float* wk   = (const float*)d_in[5];
  const float* bk   = (const float*)d_in[6];
  const float* wv   = (const float*)d_in[7];
  const float* bv   = (const float*)d_in[8];
  const float* wo   = (const float*)d_in[9];
  const float* bo   = (const float*)d_in[10];
  const float* ln2g = (const float*)d_in[11];
  const float* ln2b = (const float*)d_in[12];
  const float* w1   = (const float*)d_in[13];
  const float* w3   = (const float*)d_in[14];
  const float* w2   = (const float*)d_in[15];
  float* out = (float*)d_out;

  // ---- workspace carve (all 256B aligned) ----
  char* cur = (char*)d_ws;
  auto alloc = [&](size_t bytes) -> void* {
    void* p = (void*)cur;
    cur += (bytes + 255) & ~(size_t)255;
    return p;
  };
  _Float16* wqkv16 = (_Float16*)alloc((size_t)D_MODEL * H3 * 2);       // 6.3 MB
  _Float16* wo16   = (_Float16*)alloc((size_t)D_MODEL * D_MODEL * 2);  // 2.1 MB
  _Float16* w116   = (_Float16*)alloc((size_t)D_MODEL * HIDDEN * 2);   // 5.9 MB
  _Float16* w316   = (_Float16*)alloc((size_t)D_MODEL * HIDDEN * 2);   // 5.9 MB
  _Float16* w216   = (_Float16*)alloc((size_t)HIDDEN * D_MODEL * 2);   // 5.9 MB
  float*    bqkv   = (float*)alloc((size_t)H3 * 4);
  _Float16* hbuf   = (_Float16*)alloc((size_t)TOK * D_MODEL * 2);      // 16.8 MB (h, then h2)
  _Float16* qkvbuf = (_Float16*)alloc((size_t)TOK * H3 * 2);           // 50.3 MB (qkv, then g3)
  _Float16* attout = (_Float16*)alloc((size_t)TOK * D_MODEL * 2);      // 16.8 MB
  float*    x1     = (float*)alloc((size_t)TOK * D_MODEL * 4);         // 33.6 MB
  _Float16* g1buf  = (_Float16*)alloc((size_t)TOK * HIDDEN * 2);       // 47.2 MB (g1, then u)
  _Float16* g3buf  = qkvbuf;                                           // reuse after attention

  dim3 blk128(128), blk256(256);

  // 1) weights fp32 -> f16 (wq|wk|wv packed columnwise into [1024 x 3072])
  int nw = D_MODEL * D_MODEL;
  cvt_f32_f16<<<(nw + 255) / 256, blk256, 0, stream>>>(wq, wqkv16, D_MODEL, D_MODEL, H3, 0);
  cvt_f32_f16<<<(nw + 255) / 256, blk256, 0, stream>>>(wk, wqkv16, D_MODEL, D_MODEL, H3, D_MODEL);
  cvt_f32_f16<<<(nw + 255) / 256, blk256, 0, stream>>>(wv, wqkv16, D_MODEL, D_MODEL, H3, 2 * D_MODEL);
  cvt_f32_f16<<<(nw + 255) / 256, blk256, 0, stream>>>(wo, wo16, D_MODEL, D_MODEL, D_MODEL, 0);
  int nf = D_MODEL * HIDDEN;
  cvt_f32_f16<<<(nf + 255) / 256, blk256, 0, stream>>>(w1, w116, D_MODEL, HIDDEN, HIDDEN, 0);
  cvt_f32_f16<<<(nf + 255) / 256, blk256, 0, stream>>>(w3, w316, D_MODEL, HIDDEN, HIDDEN, 0);
  cvt_f32_f16<<<(nf + 255) / 256, blk256, 0, stream>>>(w2, w216, HIDDEN, D_MODEL, D_MODEL, 0);
  // packed QKV bias
  hipMemcpyAsync(bqkv,               bq, D_MODEL * 4, hipMemcpyDeviceToDevice, stream);
  hipMemcpyAsync(bqkv + D_MODEL,     bk, D_MODEL * 4, hipMemcpyDeviceToDevice, stream);
  hipMemcpyAsync(bqkv + 2 * D_MODEL, bv, D_MODEL * 4, hipMemcpyDeviceToDevice, stream);

  // 2) LN1
  ln_f16<<<TOK, blk256, 0, stream>>>(x, ln1g, ln1b, hbuf);

  // 3) QKV = h @ [wq|wk|wv] + [bq|bk|bv]   (8192 x 3072 x 1024)
  gemm_f16<0><<<dim3(H3 / 64, TOK / 128), blk128, 0, stream>>>(
      hbuf, D_MODEL, wqkv16, H3, D_MODEL, bqkv, nullptr, qkvbuf, nullptr, H3);

  // 4) attention
  attn_kernel<<<dim3(SEQ / 64, BATCH * N_HEADS), blk128, 0, stream>>>(qkvbuf, attout);

  // 5) x1 = x + attout @ wo + bo   (fp32 out)
  gemm_f16<1><<<dim3(D_MODEL / 64, TOK / 128), blk128, 0, stream>>>(
      attout, D_MODEL, wo16, D_MODEL, D_MODEL, bo, x, nullptr, x1, D_MODEL);

  // 6) LN2
  ln_f16<<<TOK, blk256, 0, stream>>>(x1, ln2g, ln2b, hbuf);

  // 7) g1 = h2 @ w1 ; g3 = h2 @ w3   (8192 x 2880 x 1024)
  gemm_f16<0><<<dim3(HIDDEN / 64, TOK / 128), blk128, 0, stream>>>(
      hbuf, D_MODEL, w116, HIDDEN, D_MODEL, nullptr, nullptr, g1buf, nullptr, HIDDEN);
  gemm_f16<0><<<dim3(HIDDEN / 64, TOK / 128), blk128, 0, stream>>>(
      hbuf, D_MODEL, w316, HIDDEN, D_MODEL, nullptr, nullptr, g3buf, nullptr, HIDDEN);

  // 8) u = silu(g1) * g3   (in place into g1buf)
  int nu = TOK * HIDDEN;
  swiglu_f16<<<(nu + 255) / 256, blk256, 0, stream>>>(g1buf, g3buf, nu);

  // 9) out = x1 + u @ w2   (8192 x 1024 x 2880, fp32 out)
  gemm_f16<1><<<dim3(D_MODEL / 64, TOK / 128), blk128, 0, stream>>>(
      g1buf, HIDDEN, w216, D_MODEL, HIDDEN, nullptr, x1, nullptr, out, D_MODEL);
}
